// HFALoss_8864812499471
// MI455X (gfx1250) — matrix operand order repaired and checked by reading, not compiled, not run on hardware
//
#include <hip/hip_runtime.h>
#include <hip/hip_bf16.h>
#include <math.h>

typedef __attribute__((ext_vector_type(16))) __bf16 v16bf;
typedef __attribute__((ext_vector_type(8)))  float  v8f;

#define N_N 4096
#define N_C 1000
#define N_A 64
#define C_PAD 1024
#define NTILES 63   // ceil(1000/16)

// ---------------------------------------------------------------- w2[c] = ||W_c||^2
__global__ __launch_bounds__(256) void w2_kernel(const float* __restrict__ W,
                                                 float* __restrict__ w2) {
    int c = blockIdx.x * 256 + threadIdx.x;
    if (c < N_C) {
        const float* row = W + (size_t)c * N_A;
        float s = 0.f;
        #pragma unroll
        for (int a = 0; a < N_A; ++a) { float v = row[a]; s += v * v; }
        w2[c] = s;
    }
}

// ---------------------------------------------------------------- main: one block per sample n
__global__ __launch_bounds__(256) void hfa_main(const float* __restrict__ W,     // (C,A)
                                                const float* __restrict__ Y,     // (N,C)
                                                const int*   __restrict__ tgt,   // (N,)
                                                const float* __restrict__ ratio, // scalar
                                                const float* __restrict__ CV,    // (N,A,A)
                                                const float* __restrict__ w2,    // (C,)
                                                float* __restrict__ nll)         // (N,)
{
    __shared__ float sM[N_A * N_A];        // CV_n, row-major [k][a]
    __shared__ float sWk[N_A];             // W[target] row
    __shared__ float sT[8][16 * 65];       // per-wave T tile, row padded to 65
    __shared__ float sLogit[C_PAD];
    __shared__ float sTmpA[8];
    __shared__ float sTmpB[8];

    const int n    = blockIdx.x;
    const int tid  = threadIdx.x;
    const int lane = tid & 31;
    const int wave = tid >> 5;
    const int lo   = lane & 15;    // row / column index inside 16
    const int hi   = lane >> 4;    // half-wave select

    const int k = tgt[n];
    const float* Mg = CV + (size_t)n * (N_A * N_A);
    for (int i = tid; i < N_A * N_A; i += 256) sM[i] = Mg[i];
    if (tid < N_A) sWk[tid] = W[(size_t)k * N_A + tid];
    for (int i = tid; i < C_PAD; i += 256) sLogit[i] = -INFINITY;
    __syncthreads();

    const float w2k = w2[k];
    const float rh  = 0.5f * ratio[0];

    // ---- Build B-operand registers once: B[k][a] = CV_n[k][a], bf16.
    // 16-bit 32x16 B layout: lane holds col a = nt*16+lo, K = 32*kb + {8*hi+e (e 0..7), 16+8*hi+(e-8) (e 8..15)}.
    v16bf Breg[2][4];
    #pragma unroll
    for (int kb = 0; kb < 2; ++kb) {
        const int kbase = 32 * kb + 8 * hi;
        #pragma unroll
        for (int nt = 0; nt < 4; ++nt) {
            const int a = nt * 16 + lo;
            v16bf b;
            #pragma unroll
            for (int e = 0; e < 16; ++e) b[e] = (__bf16)sM[(kbase + e + (e >= 8 ? 8 : 0)) * N_A + a];
            Breg[kb][nt] = b;
        }
    }

    float* Tb = sT[wave];
    const v8f vzero = {0.f, 0.f, 0.f, 0.f, 0.f, 0.f, 0.f, 0.f};

    for (int t = wave; t < NTILES; t += 8) {
        const int c0 = t * 16;

        // ---- A operand: d rows, kept in f32 regs for the later row-dot.
        // 16-bit 16x32 A layout: lane = row (c0+lo),
        // K runs {8*hi .. +7} (e 0..7) and {16+8*hi .. +7} (e 8..15), +32*kb.
        int cA = c0 + lo; if (cA > N_C - 1) cA = N_C - 1;   // clamp tail (rows >=1000 discarded later)
        const float* Wc = W + (size_t)cA * N_A;
        float d32[32];                 // this lane's 32 d-values (f32)
        v16bf Areg[2];
        #pragma unroll
        for (int kb = 0; kb < 2; ++kb) {
            const int koff = 32 * kb + 8 * hi;
            #pragma unroll
            for (int e = 0; e < 8; ++e)
                d32[kb * 16 + e]     = Wc[koff + e]      - sWk[koff + e];
            #pragma unroll
            for (int e = 0; e < 8; ++e)
                d32[kb * 16 + 8 + e] = Wc[koff + 16 + e] - sWk[koff + 16 + e];
            v16bf av;
            #pragma unroll
            for (int e = 0; e < 16; ++e) av[e] = (__bf16)d32[kb * 16 + e];
            Areg[kb] = av;
        }

        // ---- T tile (16x64) = d_tile @ CV_n : 2 K-steps x 4 N-tiles of WMMA bf16
        v8f acc[4] = {vzero, vzero, vzero, vzero};
        #pragma unroll
        for (int kb = 0; kb < 2; ++kb)
            #pragma unroll
            for (int nt = 0; nt < 4; ++nt)
                acc[nt] = __builtin_amdgcn_wmma_f32_16x16x32_bf16(
                    false, Areg[kb], false, Breg[kb][nt],
                    (short)0, acc[nt], false, false);

        // ---- Spill T to LDS (C/D layout: VGPR r -> row r+8*hi, col nt*16+lo)
        #pragma unroll
        for (int nt = 0; nt < 4; ++nt)
            #pragma unroll
            for (int r = 0; r < 8; ++r)
                Tb[(r + 8 * hi) * 65 + nt * 16 + lo] = acc[nt][r];
        asm volatile("s_wait_dscnt 0x0" ::: "memory");  // in-wave LDS RAW

        // ---- row-dot: lane handles row lo; its d32[] covers K-sets
        // {8*hi..+7, 16+8*hi..+7, +32} - the two half-waves cover all 64 cols.
        const int c = c0 + lo;
        float s = 0.f;
        if (c < N_C) {
            const float* Tr = Tb + lo * 65;
            #pragma unroll
            for (int kb = 0; kb < 2; ++kb) {
                const int koff = 32 * kb + 8 * hi;
                #pragma unroll
                for (int e = 0; e < 8; ++e)
                    s += Tr[koff + e]      * d32[kb * 16 + e];
                #pragma unroll
                for (int e = 0; e < 8; ++e)
                    s += Tr[koff + 16 + e] * d32[kb * 16 + 8 + e];
            }
        }
        s += __shfl_xor(s, 16, 32);
        if (hi == 0 && c < N_C)
            sLogit[c] = Y[(size_t)n * N_C + c] + rh * s + (w2[c] - w2k);
    }
    __syncthreads();

    // ---- deterministic block max
    float m = -INFINITY;
    for (int i = tid; i < C_PAD; i += 256) m = fmaxf(m, sLogit[i]);
    #pragma unroll
    for (int o = 16; o >= 1; o >>= 1) m = fmaxf(m, __shfl_xor(m, o, 32));
    if (lane == 0) sTmpA[wave] = m;
    __syncthreads();
    if (tid == 0) {
        float mm = sTmpA[0];
        for (int i = 1; i < 8; ++i) mm = fmaxf(mm, sTmpA[i]);
        sTmpA[0] = mm;
    }
    __syncthreads();
    const float bmax = sTmpA[0];

    // ---- deterministic sum of exp
    float se = 0.f;
    for (int i = tid; i < N_C; i += 256) se += expf(sLogit[i] - bmax);
    #pragma unroll
    for (int o = 16; o >= 1; o >>= 1) se += __shfl_xor(se, o, 32);
    if (lane == 0) sTmpB[wave] = se;
    __syncthreads();
    if (tid == 0) {
        float tot = 0.f;
        for (int i = 0; i < 8; ++i) tot += sTmpB[i];
        nll[n] = logf(tot) + bmax - sLogit[k];   // -log_softmax[target]
    }
}

// ---------------------------------------------------------------- final mean
__global__ __launch_bounds__(256) void loss_reduce(const float* __restrict__ nll,
                                                   float* __restrict__ out) {
    __shared__ float sTmp[8];
    const int tid = threadIdx.x, lane = tid & 31, w = tid >> 5;
    float s = 0.f;
    for (int i = tid; i < N_N; i += 256) s += nll[i];
    #pragma unroll
    for (int o = 16; o >= 1; o >>= 1) s += __shfl_xor(s, o, 32);
    if (lane == 0) sTmp[w] = s;
    __syncthreads();
    if (tid == 0) {
        float t = 0.f;
        for (int i = 0; i < 8; ++i) t += sTmp[i];
        out[0] = t / (float)N_N;
    }
}

extern "C" void kernel_launch(void* const* d_in, const int* in_sizes, int n_in,
                              void* d_out, int out_size, void* d_ws, size_t ws_size,
                              hipStream_t stream) {
    const float* W     = (const float*)d_in[0];  // weight_m (C,A)
    const float* Y     = (const float*)d_in[1];  // y (N,C)
    // d_in[2]: features (unused by reference math)
    const int*   tgt   = (const int*)d_in[3];    // target_x (N,)
    const float* ratio = (const float*)d_in[4];  // scalar
    const float* CV    = (const float*)d_in[5];  // final_conv (N,A,A)
    // d_in[6]: class_num (compile-time constant 1000)

    float* out = (float*)d_out;                  // [loss, y...]
    float* w2  = (float*)d_ws;                   // 1024 floats
    float* nll = (float*)d_ws + 1024;            // 4096 floats

    w2_kernel<<<(N_C + 255) / 256, 256, 0, stream>>>(W, w2);
    hfa_main<<<N_N, 256, 0, stream>>>(W, Y, tgt, ratio, CV, w2, nll);
    loss_reduce<<<1, 256, 0, stream>>>(nll, out);
    hipMemcpyAsync(out + 1, Y, (size_t)N_N * N_C * sizeof(float),
                   hipMemcpyDeviceToDevice, stream);
}
